// SingleStreamAttention_27882927686089
// MI455X (gfx1250) — matrix operand order repaired and checked by reading, not compiled, not run on hardware
//
#include <hip/hip_runtime.h>
#include <hip/hip_bf16.h>

typedef __attribute__((ext_vector_type(16))) _Float16 v16h;
typedef __attribute__((ext_vector_type(8)))  _Float16 v8h;
typedef __attribute__((ext_vector_type(4)))  _Float16 v4h;
typedef __attribute__((ext_vector_type(8)))  float    v8f;
typedef __attribute__((ext_vector_type(4)))  float    v4f;

#define WMMA_F32_F16(a,b,c) \
  __builtin_amdgcn_wmma_f32_16x16x32_f16(false,(a),false,(b),(short)0,(c),false,false)

// ---------------------------------------------------------------------------
// CDNA5 async global->LDS copy (ASYNCcnt-tracked, bypasses VGPRs).
// ---------------------------------------------------------------------------
__device__ __forceinline__ unsigned lds_off32(const void* p) {
  return (unsigned)(unsigned long long)(uintptr_t)p;
}
__device__ __forceinline__ void async_b128(unsigned ldsaddr, const void* g) {
  asm volatile("global_load_async_to_lds_b128 %0, %1, off"
               :: "v"(ldsaddr), "v"((unsigned long long)(uintptr_t)g)
               : "memory");
}
__device__ __forceinline__ void wait_async0() {
  asm volatile("s_wait_asynccnt 0x0" ::: "memory");
}
__device__ __forceinline__ void wait_async6() {
  asm volatile("s_wait_asynccnt 0x6" ::: "memory");
}

// ---------------------------------------------------------------------------
// WMMA fragment loaders (CDNA5 16x16x32 f16 layouts, ISA 7.12.2)
// ---------------------------------------------------------------------------
__device__ __forceinline__ v16h load_a_frag(const _Float16* base, int m, int ld,
                                            int kc, int lane) {
  const _Float16* p = base + (size_t)(m + (lane & 15)) * ld + kc + ((lane >> 4) << 3);
  v8h lo = *(const v8h*)p;
  v8h hi = *(const v8h*)(p + 16);
  v16h r;
#pragma unroll
  for (int i = 0; i < 8; ++i) { r[i] = lo[i]; r[i + 8] = hi[i]; }
  return r;
}
__device__ __forceinline__ v16h load_b_frag(const _Float16* base, int n, int ld,
                                            int kc, int lane) {
  const _Float16* p = base + (size_t)(n + (lane & 15)) * ld + kc + ((lane >> 4) << 3);
  v8h lo = *(const v8h*)p;
  v8h hi = *(const v8h*)(p + 16);
  v16h r;
#pragma unroll
  for (int i = 0; i < 8; ++i) { r[i] = lo[i]; r[i + 8] = hi[i]; }
  return r;
}

// ---------------------------------------------------------------------------
// f32 -> f16 conversion (4 elements / thread)
// ---------------------------------------------------------------------------
__global__ __launch_bounds__(256) void cvt_f32_f16(const float* __restrict__ src,
                                                   _Float16* __restrict__ dst, int n) {
  int i = (blockIdx.x * 256 + threadIdx.x) * 4;
  if (i < n) {
    v4f v = *(const v4f*)(src + i);
    v4h h = {(_Float16)v[0], (_Float16)v[1], (_Float16)v[2], (_Float16)v[3]};
    *(v4h*)(dst + i) = h;
  }
}

// ---------------------------------------------------------------------------
// Q projection + per-head LayerNorm.  Tile 128(M) x 64(N), 2x2 reg blocking,
// double-buffered LDS fed by async-to-LDS copies.
// ---------------------------------------------------------------------------
__global__ __launch_bounds__(256) void q_gemm_ln(
    const _Float16* __restrict__ X, const _Float16* __restrict__ Wq,
    const float* __restrict__ qb, const float* __restrict__ qnw,
    const float* __restrict__ qnb, _Float16* __restrict__ Qout) {
  __shared__ __align__(16) char ldsbuf[55296];
  const int tid = threadIdx.x, wave = tid >> 5, lane = tid & 31;
  const int mbase = blockIdx.x * 128, nbase = blockIdx.y * 64;
  const int mb = (wave & 3) * 32, nb = (wave >> 2) * 32;
  const int hl = lane >> 4, ln = lane & 15;

  auto Abuf = [&](int buf) -> _Float16* { return (_Float16*)(ldsbuf + buf * 18432); };
  auto Wbuf = [&](int buf) -> _Float16* { return (_Float16*)(ldsbuf + 36864 + buf * 9216); };

  auto issue = [&](int buf, int kb) {
    _Float16* Ab = Abuf(buf);
    _Float16* Wb = Wbuf(buf);
#pragma unroll
    for (int i = 0; i < 4; ++i) {   // A: 128x64 halves = 1024 x 16B chunks
      int id = tid + i * 256; int r = id >> 3; int c = (id & 7) * 8;
      async_b128(lds_off32(Ab + r * 72 + c),
                 X + (size_t)(mbase + r) * 1024 + kb + c);
    }
#pragma unroll
    for (int i = 0; i < 2; ++i) {   // W: 64x64 halves = 512 x 16B chunks
      int id = tid + i * 256; int r = id >> 3; int c = (id & 7) * 8;
      async_b128(lds_off32(Wb + r * 72 + c),
                 Wq + (size_t)(nbase + r) * 1024 + kb + c);
    }
  };

  v8f acc00 = {}, acc01 = {}, acc10 = {}, acc11 = {};
  issue(0, 0);
  const int NK = 1024 / 64;
  for (int it = 0; it < NK; ++it) {
    int cur = it & 1;
    if (it + 1 < NK) { issue(cur ^ 1, (it + 1) * 64); wait_async6(); }
    else wait_async0();
    __syncthreads();
    const _Float16* Ac = Abuf(cur);
    const _Float16* Wc = Wbuf(cur);
#pragma unroll
    for (int kc = 0; kc < 64; kc += 32) {
      v16h a0 = load_a_frag(Ac, mb, 72, kc, lane);
      v16h a1 = load_a_frag(Ac, mb + 16, 72, kc, lane);
      v16h b0 = load_b_frag(Wc, nb, 72, kc, lane);
      v16h b1 = load_b_frag(Wc, nb + 16, 72, kc, lane);
      acc00 = WMMA_F32_F16(a0, b0, acc00);
      acc01 = WMMA_F32_F16(a0, b1, acc01);
      acc10 = WMMA_F32_F16(a1, b0, acc10);
      acc11 = WMMA_F32_F16(a1, b1, acc11);
    }
    __syncthreads();
  }
  // epilog: stage (+bias) to LDS f32, fused per-head LayerNorm
  float* ldsC = (float*)ldsbuf;  // [128][68] = 34816 B (aliases tile buffers)
  float bb0 = qb[nbase + nb + ln];
  float bb1 = qb[nbase + nb + 16 + ln];
#pragma unroll
  for (int i = 0; i < 8; ++i) {
    int m0 = mb + i + 8 * hl, m1 = m0 + 16;
    ldsC[m0 * 68 + nb + ln]      = acc00[i] + bb0;
    ldsC[m0 * 68 + nb + 16 + ln] = acc01[i] + bb1;
    ldsC[m1 * 68 + nb + ln]      = acc10[i] + bb0;
    ldsC[m1 * 68 + nb + 16 + ln] = acc11[i] + bb1;
  }
  __syncthreads();
  if (tid < 128) {
    const float* row = ldsC + tid * 68;
    float mean = 0.f;
    for (int d = 0; d < 64; ++d) mean += row[d];
    mean *= (1.f / 64.f);
    float var = 0.f;
    for (int d = 0; d < 64; ++d) { float t = row[d] - mean; var += t * t; }
    var *= (1.f / 64.f);
    float rstd = rsqrtf(var + 1e-6f);
    int gr = mbase + tid;
    int bidx = gr / 1560, n = gr % 1560, h = blockIdx.y;
    _Float16* dst = Qout + (((size_t)bidx * 16 + h) * 1560 + n) * 64;
    for (int d = 0; d < 64; ++d)
      dst[d] = (_Float16)((row[d] - mean) * rstd * qnw[d] + qnb[d]);
  }
}

// ---------------------------------------------------------------------------
// KV projection; K rows LayerNormed (eps 1e-5), V passed through.
// ---------------------------------------------------------------------------
__global__ __launch_bounds__(256) void kv_gemm_ln(
    const _Float16* __restrict__ E, const _Float16* __restrict__ Wkv,
    const float* __restrict__ kvb, const float* __restrict__ knw,
    const float* __restrict__ knb, _Float16* __restrict__ Kout,
    _Float16* __restrict__ Vout) {
  __shared__ __align__(16) char ldsbuf[55296];
  const int tid = threadIdx.x, wave = tid >> 5, lane = tid & 31;
  const int mbase = blockIdx.x * 128, nbase = blockIdx.y * 64;
  const int mb = (wave & 3) * 32, nb = (wave >> 2) * 32;
  const int hl = lane >> 4, ln = lane & 15;

  auto Abuf = [&](int buf) -> _Float16* { return (_Float16*)(ldsbuf + buf * 18432); };
  auto Wbuf = [&](int buf) -> _Float16* { return (_Float16*)(ldsbuf + 36864 + buf * 9216); };

  auto issue = [&](int buf, int kb) {
    _Float16* Ab = Abuf(buf);
    _Float16* Wb = Wbuf(buf);
#pragma unroll
    for (int i = 0; i < 4; ++i) {
      int id = tid + i * 256; int r = id >> 3; int c = (id & 7) * 8;
      async_b128(lds_off32(Ab + r * 72 + c),
                 E + (size_t)(mbase + r) * 768 + kb + c);
    }
#pragma unroll
    for (int i = 0; i < 2; ++i) {
      int id = tid + i * 256; int r = id >> 3; int c = (id & 7) * 8;
      async_b128(lds_off32(Wb + r * 72 + c),
                 Wkv + (size_t)(nbase + r) * 768 + kb + c);
    }
  };

  v8f acc00 = {}, acc01 = {}, acc10 = {}, acc11 = {};
  issue(0, 0);
  const int NK = 768 / 64;
  for (int it = 0; it < NK; ++it) {
    int cur = it & 1;
    if (it + 1 < NK) { issue(cur ^ 1, (it + 1) * 64); wait_async6(); }
    else wait_async0();
    __syncthreads();
    const _Float16* Ac = Abuf(cur);
    const _Float16* Wc = Wbuf(cur);
#pragma unroll
    for (int kc = 0; kc < 64; kc += 32) {
      v16h a0 = load_a_frag(Ac, mb, 72, kc, lane);
      v16h a1 = load_a_frag(Ac, mb + 16, 72, kc, lane);
      v16h b0 = load_b_frag(Wc, nb, 72, kc, lane);
      v16h b1 = load_b_frag(Wc, nb + 16, 72, kc, lane);
      acc00 = WMMA_F32_F16(a0, b0, acc00);
      acc01 = WMMA_F32_F16(a0, b1, acc01);
      acc10 = WMMA_F32_F16(a1, b0, acc10);
      acc11 = WMMA_F32_F16(a1, b1, acc11);
    }
    __syncthreads();
  }
  float* ldsC = (float*)ldsbuf;  // [128][68]
  float bb0 = kvb[nbase + nb + ln];
  float bb1 = kvb[nbase + nb + 16 + ln];
#pragma unroll
  for (int i = 0; i < 8; ++i) {
    int m0 = mb + i + 8 * hl, m1 = m0 + 16;
    ldsC[m0 * 68 + nb + ln]      = acc00[i] + bb0;
    ldsC[m0 * 68 + nb + 16 + ln] = acc01[i] + bb1;
    ldsC[m1 * 68 + nb + ln]      = acc10[i] + bb0;
    ldsC[m1 * 68 + nb + 16 + ln] = acc11[i] + bb1;
  }
  __syncthreads();
  if (tid < 128) {
    const float* row = ldsC + tid * 68;
    int gr = mbase + tid;
    int bidx = gr >> 8, na = gr & 255;
    int kvsel = blockIdx.y >> 4, h = blockIdx.y & 15;
    _Float16* dst = (kvsel == 0 ? Kout : Vout) +
                    (((size_t)bidx * 16 + h) * 256 + na) * 64;
    if (kvsel == 0) {
      float mean = 0.f;
      for (int d = 0; d < 64; ++d) mean += row[d];
      mean *= (1.f / 64.f);
      float var = 0.f;
      for (int d = 0; d < 64; ++d) { float t = row[d] - mean; var += t * t; }
      var *= (1.f / 64.f);
      float rstd = rsqrtf(var + 1e-5f);
      for (int d = 0; d < 64; ++d)
        dst[d] = (_Float16)((row[d] - mean) * rstd * knw[d] + knb[d]);
    } else {
      for (int d = 0; d < 64; ++d) dst[d] = (_Float16)row[d];
    }
  }
}

// ---------------------------------------------------------------------------
// Attention per (b,h): 32-query block vs all 256 keys.  K/Q staged via
// async-to-LDS; S = QK^T (WMMA), softmax, O = P V (WMMA).
// ---------------------------------------------------------------------------
__global__ __launch_bounds__(256) void attn_kernel(
    const _Float16* __restrict__ Q, const _Float16* __restrict__ Kf,
    const _Float16* __restrict__ Vf, _Float16* __restrict__ O) {
  extern __shared__ __align__(16) char sm[];
  _Float16* Kl = (_Float16*)sm;              // [256][72]
  _Float16* VT = Kl + 256 * 72;              // [64][264]  (V transposed)
  _Float16* Qt = VT + 64 * 264;              // [32][72]
  float*    S  = (float*)(Qt + 32 * 72);     // [32][260]
  _Float16* P  = (_Float16*)(S + 32 * 260);  // [32][264]
  const int tid = threadIdx.x, wave = tid >> 5, lane = tid & 31;
  const int hl = lane >> 4, ln = lane & 15;
  const int b = blockIdx.y >> 4, h = blockIdx.y & 15;
  const int mbase = blockIdx.x * 32;
  const _Float16* Kg = Kf + ((size_t)b * 16 + h) * 256 * 64;
  const _Float16* Vg = Vf + ((size_t)b * 16 + h) * 256 * 64;
  const _Float16* Qg = Q  + ((size_t)b * 16 + h) * 1560 * 64;
  // async stage K rows (8 x b128 per thread)
#pragma unroll
  for (int i = 0; i < 8; ++i) {
    int id = tid + i * 256; int r = id >> 3; int c = (id & 7) * 8;
    async_b128(lds_off32(Kl + r * 72 + c), Kg + (size_t)r * 64 + c);
  }
  // async stage 32-row Q tile (clamp tail rows; garbage rows never stored)
  {
    int r = tid >> 3; int c = (tid & 7) * 8;
    int gr = mbase + r; if (gr > 1559) gr = 1559;
    async_b128(lds_off32(Qt + r * 72 + c), Qg + (size_t)gr * 64 + c);
  }
  // V transposed into LDS via regular path (async cannot reshuffle)
#pragma unroll
  for (int i = 0; i < 8; ++i) {
    int id = tid + i * 256; int na = id >> 3; int d0 = (id & 7) * 8;
    v8h v = *(const v8h*)(Vg + (size_t)na * 64 + d0);
#pragma unroll
    for (int j = 0; j < 8; ++j) VT[(d0 + j) * 264 + na] = v[j];
  }
  wait_async0();
  __syncthreads();
  // S = (Q K^T) * scale
  {
    int mi = wave & 1; int nbas = (wave >> 1) * 64;
    v16h a0 = load_a_frag(Qt, mi * 16, 72, 0, lane);
    v16h a1 = load_a_frag(Qt, mi * 16, 72, 32, lane);
#pragma unroll
    for (int j = 0; j < 4; ++j) {
      int n0 = nbas + j * 16;
      v8f acc = {};
      v16h b0 = load_b_frag(Kl, n0, 72, 0, lane);
      acc = WMMA_F32_F16(a0, b0, acc);
      v16h b1 = load_b_frag(Kl, n0, 72, 32, lane);
      acc = WMMA_F32_F16(a1, b1, acc);
#pragma unroll
      for (int i = 0; i < 8; ++i) {
        int m = mi * 16 + i + 8 * hl;
        S[m * 260 + n0 + ln] = acc[i] * 0.125f;
      }
    }
  }
  __syncthreads();
  // softmax rows -> P (f16)
  if (tid < 32) {
    float* row = S + tid * 260;
    float mx = -3.0e38f;
    for (int n = 0; n < 256; ++n) mx = fmaxf(mx, row[n]);
    float sum = 0.f;
    for (int n = 0; n < 256; ++n) { float e = __expf(row[n] - mx); row[n] = e; sum += e; }
    float inv = 1.0f / sum;
    _Float16* pr = P + tid * 264;
    for (int n = 0; n < 256; ++n) pr[n] = (_Float16)(row[n] * inv);
  }
  __syncthreads();
  // O = P V
  {
    int mi = wave & 1; int no = (wave >> 1) * 16;
    v8f acc = {};
#pragma unroll
    for (int kc = 0; kc < 256; kc += 32) {
      v16h a  = load_a_frag(P, mi * 16, 264, kc, lane);
      v16h bb = load_b_frag(VT, no, 264, kc, lane);
      acc = WMMA_F32_F16(a, bb, acc);
    }
#pragma unroll
    for (int i = 0; i < 8; ++i) {
      int m = mi * 16 + i + 8 * hl;
      int gr = mbase + m;
      if (gr < 1560)
        O[((size_t)b * 1560 + gr) * 1024 + h * 64 + no + ln] = (_Float16)acc[i];
    }
  }
}

// ---------------------------------------------------------------------------
// Output projection: d_out = O(f16) @ Wp^T + pb, f32 store.
// ---------------------------------------------------------------------------
__global__ __launch_bounds__(256) void proj_gemm(
    const _Float16* __restrict__ Of, const _Float16* __restrict__ Wp,
    const float* __restrict__ pb, float* __restrict__ out) {
  __shared__ __align__(16) char ldsbuf[55296];
  const int tid = threadIdx.x, wave = tid >> 5, lane = tid & 31;
  const int mbase = blockIdx.x * 128, nbase = blockIdx.y * 64;
  const int mb = (wave & 3) * 32, nb = (wave >> 2) * 32;
  const int hl = lane >> 4, ln = lane & 15;

  auto Abuf = [&](int buf) -> _Float16* { return (_Float16*)(ldsbuf + buf * 18432); };
  auto Wbuf = [&](int buf) -> _Float16* { return (_Float16*)(ldsbuf + 36864 + buf * 9216); };

  auto issue = [&](int buf, int kb) {
    _Float16* Ab = Abuf(buf);
    _Float16* Wb = Wbuf(buf);
#pragma unroll
    for (int i = 0; i < 4; ++i) {
      int id = tid + i * 256; int r = id >> 3; int c = (id & 7) * 8;
      async_b128(lds_off32(Ab + r * 72 + c),
                 Of + (size_t)(mbase + r) * 1024 + kb + c);
    }
#pragma unroll
    for (int i = 0; i < 2; ++i) {
      int id = tid + i * 256; int r = id >> 3; int c = (id & 7) * 8;
      async_b128(lds_off32(Wb + r * 72 + c),
                 Wp + (size_t)(nbase + r) * 1024 + kb + c);
    }
  };

  v8f acc00 = {}, acc01 = {}, acc10 = {}, acc11 = {};
  issue(0, 0);
  const int NK = 1024 / 64;
  for (int it = 0; it < NK; ++it) {
    int cur = it & 1;
    if (it + 1 < NK) { issue(cur ^ 1, (it + 1) * 64); wait_async6(); }
    else wait_async0();
    __syncthreads();
    const _Float16* Ac = Abuf(cur);
    const _Float16* Wc = Wbuf(cur);
#pragma unroll
    for (int kc = 0; kc < 64; kc += 32) {
      v16h a0 = load_a_frag(Ac, mb, 72, kc, lane);
      v16h a1 = load_a_frag(Ac, mb + 16, 72, kc, lane);
      v16h b0 = load_b_frag(Wc, nb, 72, kc, lane);
      v16h b1 = load_b_frag(Wc, nb + 16, 72, kc, lane);
      acc00 = WMMA_F32_F16(a0, b0, acc00);
      acc01 = WMMA_F32_F16(a0, b1, acc01);
      acc10 = WMMA_F32_F16(a1, b0, acc10);
      acc11 = WMMA_F32_F16(a1, b1, acc11);
    }
    __syncthreads();
  }
  float bb0 = pb[nbase + nb + ln];
  float bb1 = pb[nbase + nb + 16 + ln];
#pragma unroll
  for (int i = 0; i < 8; ++i) {
    int m0 = mb + i + 8 * hl, m1 = m0 + 16;
    size_t g0 = (size_t)(mbase + m0), g1 = (size_t)(mbase + m1);
    out[g0 * 1024 + nbase + nb + ln]      = acc00[i] + bb0;
    out[g0 * 1024 + nbase + nb + 16 + ln] = acc01[i] + bb1;
    out[g1 * 1024 + nbase + nb + ln]      = acc10[i] + bb0;
    out[g1 * 1024 + nbase + nb + 16 + ln] = acc11[i] + bb1;
  }
}

// ---------------------------------------------------------------------------
extern "C" void kernel_launch(void* const* d_in, const int* in_sizes, int n_in,
                              void* d_out, int out_size, void* d_ws, size_t ws_size,
                              hipStream_t stream) {
  (void)in_sizes; (void)n_in; (void)out_size; (void)ws_size;
  const float* x    = (const float*)d_in[0];   // [1][24960][1024]
  const float* enc  = (const float*)d_in[1];   // [16][256][768]
  const float* q_w  = (const float*)d_in[2];   // [1024][1024]
  const float* q_b  = (const float*)d_in[3];
  const float* kv_w = (const float*)d_in[4];   // [2048][768]
  const float* kv_b = (const float*)d_in[5];
  const float* p_w  = (const float*)d_in[6];   // [1024][1024]
  const float* p_b  = (const float*)d_in[7];
  const float* qn_w = (const float*)d_in[8];
  const float* qn_b = (const float*)d_in[9];
  const float* kn_w = (const float*)d_in[10];
  const float* kn_b = (const float*)d_in[11];
  float* out = (float*)d_out;

  char* ws = (char*)d_ws;
  size_t o = 0;
  _Float16* Wq  = (_Float16*)(ws + o); o += (size_t)1024 * 1024 * 2;
  _Float16* Wkv = (_Float16*)(ws + o); o += (size_t)2048 * 768 * 2;
  _Float16* Wp  = (_Float16*)(ws + o); o += (size_t)1024 * 1024 * 2;
  _Float16* Qf  = (_Float16*)(ws + o); o += (size_t)16 * 16 * 1560 * 64 * 2;
  _Float16* Kf  = (_Float16*)(ws + o); o += (size_t)16 * 16 * 256 * 64 * 2;
  _Float16* Vf  = (_Float16*)(ws + o); o += (size_t)16 * 16 * 256 * 64 * 2;
  _Float16* Of  = (_Float16*)(ws + o); o += (size_t)24960 * 1024 * 2;
  _Float16* Xf  = (_Float16*)(ws + o); o += (size_t)24960 * 1024 * 2;
  _Float16* Ef  = (_Float16*)(ws + o); o += (size_t)4096 * 768 * 2;

  cvt_f32_f16<<<1024,  256, 0, stream>>>(q_w,  Wq,  1024 * 1024);
  cvt_f32_f16<<<1536,  256, 0, stream>>>(kv_w, Wkv, 2048 * 768);
  cvt_f32_f16<<<1024,  256, 0, stream>>>(p_w,  Wp,  1024 * 1024);
  cvt_f32_f16<<<24960, 256, 0, stream>>>(x,    Xf,  24960 * 1024);
  cvt_f32_f16<<<3072,  256, 0, stream>>>(enc,  Ef,  4096 * 768);

  q_gemm_ln<<<dim3(195, 16), 256, 0, stream>>>(Xf, Wq, q_b, qn_w, qn_b, Qf);
  kv_gemm_ln<<<dim3(32, 32), 256, 0, stream>>>(Ef, Wkv, kv_b, kn_w, kn_b, Kf, Vf);

  const int attn_lds = 125440;  // K 36864 + VT 33792 + Qt 4608 + S 33280 + P 16896
  (void)hipFuncSetAttribute(reinterpret_cast<const void*>(attn_kernel),
                            hipFuncAttributeMaxDynamicSharedMemorySize, attn_lds);
  attn_kernel<<<dim3(49, 256), 256, attn_lds, stream>>>(Qf, Kf, Vf, Of);

  proj_gemm<<<dim3(195, 16), 256, 0, stream>>>(Of, Wp, p_b, out);
}
